// HeteroLayer_orig_23192823399226
// MI455X (gfx1250) — matrix operand order repaired and checked by reading, not compiled, not run on hardware
//
#include <hip/hip_runtime.h>

// ---------------------------------------------------------------------------
// HeteroLayer for MI455X (gfx1250, wave32, WMMA)
//
// Sizes (fixed by the reference):
//   N_WORD=50000, N_TOPIC=4000, N_DOC=20000, D=128
//   edges: ww=800000, wt=400000, wd=400000, td=160000, tt=160000
// ---------------------------------------------------------------------------

typedef __attribute__((ext_vector_type(16))) _Float16 v16h;
typedef __attribute__((ext_vector_type(8)))  float    v8f;

#define N_WORD  50000
#define N_TOPIC 4000
#define N_DOC   20000
#define DIM     128

// ---- workspace layout (bytes) ---------------------------------------------
static constexpr size_t OFF_WH   = 0;                          // 5 * 128*128 f16 = 163840 B
static constexpr size_t OFF_BUF1 = 163840;                     // 50000*128 f32
static constexpr size_t OFF_BUF2 = OFF_BUF1 + 25600000;        // 50000*128 f32 (sums)
static constexpr size_t OFF_CNTW = OFF_BUF2 + 25600000;        // 50000 f32
static constexpr size_t OFF_T1   = OFF_CNTW + 200000;          // 4000*128 f32
static constexpr size_t OFF_TF   = OFF_T1   + 2048000;         // 4000*128 f32
static constexpr size_t OFF_SWT  = OFF_TF   + 2048000;         // 4000*128 f32
static constexpr size_t OFF_CWT  = OFF_SWT  + 2048000;         // 4000 f32
static constexpr size_t OFF_STT  = OFF_CWT  + 16000;           // 4000*128 f32
static constexpr size_t OFF_CTT  = OFF_STT  + 2048000;         // 4000 f32
static constexpr size_t OFF_SWD  = OFF_CTT  + 16000;           // 20000*128 f32
static constexpr size_t OFF_CWD  = OFF_SWD  + 10240000;        // 20000 f32
static constexpr size_t OFF_STD  = OFF_CWD  + 80000;           // 20000*128 f32
static constexpr size_t OFF_CTD  = OFF_STD  + 10240000;        // 20000 f32
static constexpr size_t OFF_END  = OFF_CTD  + 80000;           // ~80.4 MB total

// ---------------------------------------------------------------------------
__global__ __launch_bounds__(256) void k_zero(float* __restrict__ p, size_t n) {
  size_t i = (size_t)blockIdx.x * blockDim.x + threadIdx.x;
  size_t stride = (size_t)gridDim.x * blockDim.x;
  for (; i < n; i += stride) p[i] = 0.0f;
}

__global__ __launch_bounds__(256) void k_f32_to_f16(const float* __restrict__ s,
                                                    _Float16* __restrict__ d, int n) {
  int i = blockIdx.x * blockDim.x + threadIdx.x;
  if (i < n) d[i] = (_Float16)s[i];
}

// ---------------------------------------------------------------------------
// Y[N,128] = rowscale(X)[N,128] @ W^T + b  with W stored f16 row-major [128,128]
// One block = 4 waves, each wave produces a 16x128 output strip via
// 4 (K-chunks) x 8 (N-tiles) v_wmma_f32_16x16x32_f16. W lives in LDS.
// If use_rowdiv, each X row r is multiplied by 1/max(rowdiv[r],1)  (mean agg).
// ---------------------------------------------------------------------------
__global__ __launch_bounds__(128) void k_gemm128(
    const float* __restrict__ X, const _Float16* __restrict__ Wh,
    const float* __restrict__ bias, const float* __restrict__ rowdiv,
    float* __restrict__ Y, int N, int use_rowdiv)
{
  __shared__ __align__(32) _Float16 sW[DIM * DIM];   // 32 KB
  // cooperative load of W (2048 uint4 = 32 KB) into LDS
  {
    const uint4* wsrc = (const uint4*)Wh;
    uint4* wdst = (uint4*)sW;
    for (int i = threadIdx.x; i < 2048; i += 128) wdst[i] = wsrc[i];
  }
  __syncthreads();

  const int wave = threadIdx.x >> 5;
  const int lane = threadIdx.x & 31;
  const int tile = blockIdx.x * 4 + wave;
  if (tile * 16 >= N) return;                        // whole wave exits together

  const int row0 = tile * 16;
  const int m    = lane & 15;
  const int hi   = lane >> 4;
  const int row  = row0 + m;

  float scale = 1.0f;
  if (use_rowdiv) scale = 1.0f / fmaxf(rowdiv[row], 1.0f);
  const float* xrow = X + (size_t)row * DIM;

  v8f acc[8] = {};

#pragma unroll
  for (int kc = 0; kc < 4; ++kc) {
    const int kbase = kc * 32;
    // A fragment (16x32 f16): half j -> K = hi*16 + g*8 + (j&7)
    v16h a;
#pragma unroll
    for (int g = 0; g < 2; ++g) {
      const int k0 = kbase + hi * 16 + g * 8;
#pragma unroll
      for (int j = 0; j < 8; ++j)
        a[g * 8 + j] = (_Float16)(xrow[k0 + j] * scale);
    }
#pragma unroll
    for (int ot = 0; ot < 8; ++ot) {
      const int o0 = ot * 16;
      const int n  = lane & 15;
      // B fragment (32x16 f16): half j -> K = hi*16 + j ; B[k][n] = W[o0+n][k]
      const _Float16* bp = &sW[(o0 + n) * DIM + kbase + hi * 16];
      v16h b = *(const v16h*)bp;                     // 32B contiguous in LDS
      acc[ot] = __builtin_amdgcn_wmma_f32_16x16x32_f16(
          false, a, false, b, (short)0, acc[ot], false, false);
    }
  }

  // epilogue: D lane holds column n = lane&15, VGPR r -> row (r + 8*hi)
  const int n = lane & 15;
#pragma unroll
  for (int ot = 0; ot < 8; ++ot) {
    const int o0 = ot * 16;
    const float bz = bias[o0 + n];
#pragma unroll
    for (int r = 0; r < 8; ++r) {
      const int mm = r + 8 * hi;
      Y[(size_t)(row0 + mm) * DIM + o0 + n] = acc[ot][r] + bz;
    }
  }
}

// ---------------------------------------------------------------------------
// Edge scatter: one wave per edge; each lane handles 4 floats of the 128-dim
// message. sum[dst] += H[src]*w ; cnt[dst] += 1.
// ---------------------------------------------------------------------------
__global__ __launch_bounds__(256) void k_agg(
    const float* __restrict__ H, const int* __restrict__ src,
    const int* __restrict__ dst, const float* __restrict__ w,
    float* __restrict__ sum, float* __restrict__ cnt, int E)
{
  const int e    = (blockIdx.x * 256 + threadIdx.x) >> 5;
  const int lane = threadIdx.x & 31;
  if (e >= E) return;
  const int   s  = src[e];
  const int   d  = dst[e];
  const float we = w[e];
  const float4 v = ((const float4*)(H + (size_t)s * DIM))[lane];
  float* sp = sum + (size_t)d * DIM + lane * 4;
  unsafeAtomicAdd(sp + 0, v.x * we);
  unsafeAtomicAdd(sp + 1, v.y * we);
  unsafeAtomicAdd(sp + 2, v.z * we);
  unsafeAtomicAdd(sp + 3, v.w * we);
  if (lane == 0) unsafeAtomicAdd(cnt + d, 1.0f);
}

// out = sa/max(ca,1) + sb/max(cb,1)   elementwise over n rows x 128
__global__ __launch_bounds__(256) void k_combine(
    const float* __restrict__ sa, const float* __restrict__ ca,
    const float* __restrict__ sb, const float* __restrict__ cb,
    float* __restrict__ out, int n)
{
  const int i = blockIdx.x * blockDim.x + threadIdx.x;
  if (i >= n * DIM) return;
  const int r = i >> 7;
  out[i] = sa[i] / fmaxf(ca[r], 1.0f) + sb[i] / fmaxf(cb[r], 1.0f);
}

// ---------------------------------------------------------------------------
extern "C" void kernel_launch(void* const* d_in, const int* in_sizes, int n_in,
                              void* d_out, int out_size, void* d_ws, size_t ws_size,
                              hipStream_t stream) {
  (void)in_sizes; (void)n_in; (void)out_size; (void)ws_size;

  const float* h_word  = (const float*)d_in[0];
  const float* h_topic = (const float*)d_in[1];
  const float* Wf[5]   = {(const float*)d_in[2], (const float*)d_in[4],
                          (const float*)d_in[6], (const float*)d_in[8],
                          (const float*)d_in[10]};                       // ww, wt, wd, td, tt
  const float* bias[5] = {(const float*)d_in[3], (const float*)d_in[5],
                          (const float*)d_in[7], (const float*)d_in[9],
                          (const float*)d_in[11]};
  const int*   esrc[5] = {(const int*)d_in[12], (const int*)d_in[15],
                          (const int*)d_in[18], (const int*)d_in[21],
                          (const int*)d_in[24]};                         // ww, wt, wd, td, tt
  const int*   edst[5] = {(const int*)d_in[13], (const int*)d_in[16],
                          (const int*)d_in[19], (const int*)d_in[22],
                          (const int*)d_in[25]};
  const float* ew[5]   = {(const float*)d_in[14], (const float*)d_in[17],
                          (const float*)d_in[20], (const float*)d_in[23],
                          (const float*)d_in[26]};
  const int E[5] = {800000, 400000, 400000, 160000, 160000};

  char* ws = (char*)d_ws;
  _Float16* WH       = (_Float16*)(ws + OFF_WH);
  float* buf1        = (float*)(ws + OFF_BUF1);
  float* buf2        = (float*)(ws + OFF_BUF2);
  float* cntw        = (float*)(ws + OFF_CNTW);
  float* topic_t1    = (float*)(ws + OFF_T1);
  float* topic_final = (float*)(ws + OFF_TF);
  float* sum_wt      = (float*)(ws + OFF_SWT);
  float* cnt_wt      = (float*)(ws + OFF_CWT);
  float* sum_tt      = (float*)(ws + OFF_STT);
  float* cnt_tt      = (float*)(ws + OFF_CTT);
  float* sum_wd      = (float*)(ws + OFF_SWD);
  float* cnt_wd      = (float*)(ws + OFF_CWD);
  float* sum_td      = (float*)(ws + OFF_STD);
  float* cnt_td      = (float*)(ws + OFF_CTD);

  float* out_word  = (float*)d_out;                   // [50000,128]
  float* out_topic = out_word + (size_t)N_WORD * DIM; // [4000,128]
  float* out_doc   = out_topic + (size_t)N_TOPIC * DIM;

  // 1) zero all accumulators (buf2 .. end of ws region)
  {
    size_t nz = (OFF_END - OFF_BUF2) / 4;
    k_zero<<<4096, 256, 0, stream>>>(buf2, nz);
  }
  // 2) convert the 5 weight matrices to f16
  for (int i = 0; i < 5; ++i)
    k_f32_to_f16<<<(DIM * DIM + 255) / 256, 256, 0, stream>>>(Wf[i], WH + (size_t)i * DIM * DIM, DIM * DIM);

  auto gemm = [&](const float* X, int widx, const float* rowdiv, float* Y, int N) {
    int tiles  = N / 16;
    int blocks = (tiles + 3) / 4;
    k_gemm128<<<blocks, 128, 0, stream>>>(X, WH + (size_t)widx * DIM * DIM,
                                          bias[widx], rowdiv, Y, N, rowdiv ? 1 : 0);
  };
  auto agg = [&](const float* H, int eidx, float* sum, float* cnt) {
    int blocks = (E[eidx] * 32 + 255) / 256;
    k_agg<<<blocks, 256, 0, stream>>>(H, esrc[eidx], edst[eidx], ew[eidx], sum, cnt, E[eidx]);
  };

  // 3) phase 1: wh = h_word @ W_ww^T + b_ww ; mean-agg over ww edges
  gemm(h_word, 0, nullptr, buf1, N_WORD);
  agg(buf1, 0, buf2, cntw);                                  // buf2 = sums, cntw = degrees

  // 4) phase 2 linears (mean normalization folded into A-load of next GEMM)
  gemm(buf2, 1, cntw, buf1, N_WORD);                         // linear(h_word1, W_wt)
  gemm(buf1, 2, nullptr, out_word, N_WORD);                  // word_final -> output
  gemm(h_topic, 3, nullptr, topic_t1, N_TOPIC);              // linear(h_topic, W_td)
  gemm(topic_t1, 4, nullptr, topic_final, N_TOPIC);          // topic_final

  // 5) phase 2 aggregations
  agg(out_word,    1, sum_wt, cnt_wt);                       // wt -> topics
  agg(topic_final, 4, sum_tt, cnt_tt);                       // tt -> topics
  agg(out_word,    2, sum_wd, cnt_wd);                       // wd -> docs
  agg(topic_final, 3, sum_td, cnt_td);                       // td -> docs

  // 6) combine means into outputs
  k_combine<<<(N_TOPIC * DIM + 255) / 256, 256, 0, stream>>>(sum_wt, cnt_wt, sum_tt, cnt_tt,
                                                             out_topic, N_TOPIC);
  k_combine<<<(N_DOC * DIM + 255) / 256, 256, 0, stream>>>(sum_wd, cnt_wd, sum_td, cnt_td,
                                                           out_doc, N_DOC);
}
